// LSTMSeq2Seq_88536455840157
// MI455X (gfx1250) — matrix operand order repaired and checked by reading, not compiled, not run on hardware
//
#include <hip/hip_runtime.h>

typedef __bf16 bf16_t;
typedef float v8f __attribute__((ext_vector_type(8)));
typedef bf16_t v16bf __attribute__((ext_vector_type(16)));
typedef unsigned int u32;
typedef u32 v4u __attribute__((ext_vector_type(4)));

union FragU { v4u q[2]; v16bf v; };

#define BATCH 64
#define HDIM 512
#define MAX_KTOT 1536

__device__ inline float sigm(float x) { return 1.0f / (1.0f + __expf(-x)); }

// ---------------------------------------------------------------------------
// Pack W^T (K x N) into per-WMMA-tile fragment order (bf16).
// Tile (kt, nt): element index = ((kt*NT + nt)*32 + lane)*16 + s
//   n = nt*16 + (lane & 15);  k = kt*32 + (lane>>4)*8 + (s&7) + (s>>3)*16
// Row n of weight: k < Din -> Wih[n][k], else Whh[n][k-Din].
// ---------------------------------------------------------------------------
__global__ void pack_w_kernel(const float* __restrict__ Wih,
                              const float* __restrict__ Whh,
                              bf16_t* __restrict__ out,
                              int Din, int Ktot, int Nrows) {
  const int NT = Nrows >> 4;
  const size_t total = (size_t)(Ktot >> 5) * NT * 512;
  for (size_t idx = (size_t)blockIdx.x * blockDim.x + threadIdx.x; idx < total;
       idx += (size_t)gridDim.x * blockDim.x) {
    const int s = (int)(idx & 15);
    const int lane = (int)((idx >> 4) & 31);
    const size_t tileid = idx >> 9;
    const int nt = (int)(tileid % NT);
    const int kt = (int)(tileid / NT);
    const int n = (nt << 4) + (lane & 15);
    const int k = (kt << 5) + ((lane >> 4) << 3) + (s & 7) + ((s >> 3) << 4);
    const float v = (k < Din) ? Wih[(size_t)n * Din + k]
                              : Whh[(size_t)n * HDIM + (k - Din)];
    out[idx] = (bf16_t)v;
  }
}

// src [B][T][F] f32 -> dst [T][B][F] bf16 (time-major)
__global__ void to_tm_bf16(const float* __restrict__ src, bf16_t* __restrict__ dst,
                           int B_, int T_, int F_) {
  const size_t total = (size_t)B_ * T_ * F_;
  for (size_t idx = (size_t)blockIdx.x * blockDim.x + threadIdx.x; idx < total;
       idx += (size_t)gridDim.x * blockDim.x) {
    const int f = (int)(idx % F_);
    const int t = (int)((idx / F_) % T_);
    const int b = (int)(idx / ((size_t)F_ * T_));
    dst[((size_t)t * B_ + b) * F_ + f] = (bf16_t)src[idx];
  }
}

// Decoder input: d0[t][b][0..127] = tgt[b][t][:]; d0[t][b][128..639] = sc_emb[scen[b]][:]
__global__ void build_dec_in(const float* __restrict__ tgt, const int* __restrict__ scen,
                             const float* __restrict__ sc_emb, bf16_t* __restrict__ d0,
                             int T_) {
  const size_t total = (size_t)T_ * BATCH * 640;
  for (size_t idx = (size_t)blockIdx.x * blockDim.x + threadIdx.x; idx < total;
       idx += (size_t)gridDim.x * blockDim.x) {
    const int c = (int)(idx % 640);
    const int b = (int)((idx / 640) % BATCH);
    const int t = (int)(idx / (640 * BATCH));
    float v;
    if (c < 128) v = tgt[((size_t)b * T_ + t) * 128 + c];
    else         v = sc_emb[(size_t)scen[b] * HDIM + (c - 128)];
    d0[idx] = (bf16_t)v;
  }
}

// Bridge: h0[l][b][h] = tanh(sum_d hcat[l][b][d]*hW[h][d] + hb[h]); same for c.
__global__ void bridge_kernel(const float* __restrict__ hcat, const float* __restrict__ ccat,
                              const float* __restrict__ hW, const float* __restrict__ hb,
                              const float* __restrict__ cW, const float* __restrict__ cb,
                              float* __restrict__ h0, float* __restrict__ c0) {
  const size_t total = (size_t)4 * BATCH * HDIM;
  for (size_t idx = (size_t)blockIdx.x * blockDim.x + threadIdx.x; idx < total;
       idx += (size_t)gridDim.x * blockDim.x) {
    const int hc = (int)(idx % HDIM);
    const int b = (int)((idx / HDIM) % BATCH);
    const int l = (int)(idx / ((size_t)HDIM * BATCH));
    const float* hrow = hcat + ((size_t)l * BATCH + b) * 1024;
    const float* crow = ccat + ((size_t)l * BATCH + b) * 1024;
    float ah = hb[hc], ac = cb[hc];
    for (int d = 0; d < 1024; ++d) {
      ah += hrow[d] * hW[(size_t)hc * 1024 + d];
      ac += crow[d] * cW[(size_t)hc * 1024 + d];
    }
    h0[idx] = tanhf(ah);
    c0[idx] = tanhf(ac);
  }
}

// ---------------------------------------------------------------------------
// Persistent fused LSTM layer kernel.
// Grid: (4 batch-chunks, ndir). Block: 512 threads = 16 waves.
// Each block owns 16 batch rows; loops over T steps.
// Per step: G = [x_t ; h] @ W^T + b via v_wmma_f32_16x16x32_bf16, then cell.
// A = [x_t ; h] lives in LDS (16 x Ktot bf16); c lives in registers.
// B fragments double-grouped: group 1 (4 frags) loaded, then its WMMAs issue
// while group 2's 8 global_load_b128 are in flight -> deep load pipelining.
// ---------------------------------------------------------------------------
__global__ __launch_bounds__(512)
void lstm_layer_kernel(const bf16_t* __restrict__ xT, int T, int Din,
                       const bf16_t* __restrict__ Wpk0, const bf16_t* __restrict__ Wpk1,
                       const float* __restrict__ bias0, const float* __restrict__ bias1,
                       bf16_t* __restrict__ y, int yStride, int bidir,
                       const float* __restrict__ h0, const float* __restrict__ c0,
                       float* __restrict__ hT, float* __restrict__ cT, int hcStride) {
  __shared__ alignas(16) bf16_t Als[16 * MAX_KTOT];
  const int Ktot = Din + HDIM;
  const int nk = Ktot >> 5;
  const int mchunk = blockIdx.x;                  // 0..3
  const int dir = bidir ? blockIdx.y : 0;
  const bool reverse = bidir && (dir == 1);
  const bf16_t* Wpk = dir ? Wpk1 : Wpk0;
  const float* bias = dir ? bias1 : bias0;
  const int ycol0 = bidir ? dir * HDIM : 0;

  const int tid = threadIdx.x;
  const int lane = tid & 31;
  const int wave = tid >> 5;                      // 0..15
  const int hbase = wave << 1;                    // h-tile base (2 per wave)
  const int nlo = lane & 15;
  const int khi = lane >> 4;                      // 0/1
  const int mlB = khi << 3;                       // C/D layout: m_local base

  // Bias accum templates (n depends only on lane & tile) -- built once.
  v8f binit[8];
#pragma unroll
  for (int g = 0; g < 4; ++g)
#pragma unroll
    for (int j = 0; j < 2; ++j) {
      const float bv = bias[((g * 32 + hbase + j) << 4) + nlo];
      v8f a;
#pragma unroll
      for (int r = 0; r < 8; ++r) a[r] = bv;
      binit[g * 2 + j] = a;
    }

  // c in registers; h0 into LDS h-region (bf16)
  float creg[2][8];
#pragma unroll
  for (int j = 0; j < 2; ++j) {
    const int n = ((hbase + j) << 4) + nlo;
#pragma unroll
    for (int r = 0; r < 8; ++r) {
      const int ml = mlB + r;
      const int mg = (mchunk << 4) + ml;
      const float cv = c0 ? c0[(size_t)mg * HDIM + n] : 0.0f;
      const float hv = h0 ? h0[(size_t)mg * HDIM + n] : 0.0f;
      creg[j][r] = cv;
      Als[ml * Ktot + Din + n] = (bf16_t)hv;
    }
  }

  // Fixed per-thread fragment bases.
  const bf16_t* abase = Als + nlo * Ktot + (khi << 3);              // LDS A rows
  const bf16_t* bbase0 = Wpk + ((size_t)hbase << 9) + (lane << 4);  // tile (0, hbase)

  // Preload x for first step
  {
    const int t0 = reverse ? (T - 1) : 0;
    const int half = Din >> 1;
    const u32* xs = (const u32*)(xT + (size_t)t0 * BATCH * Din) + (size_t)(mchunk * 16) * half;
    for (int idx = tid; idx < 16 * half; idx += 512) {
      const int m = idx / half;
      const int o = idx - m * half;
      *((u32*)(Als + m * Ktot) + o) = xs[(size_t)m * half + o];
    }
  }
  __syncthreads();

  for (int step = 0; step < T; ++step) {
    const int t = reverse ? (T - 1 - step) : step;

    // Accumulators start at bias
    v8f acc[8];
#pragma unroll
    for (int i = 0; i < 8; ++i) acc[i] = binit[i];

    // GEMM: [16 x Ktot] x [Ktot x 2048] -> 8 tiles/wave.
    // Tile i = g*2+j at byte-immediate offset ((i>>1)*32 + (i&1)) * 1024.
    const bf16_t* bb = bbase0;
#pragma unroll 1
    for (int kt = 0; kt < nk; ++kt) {
      FragU ua;
      ua.q[0] = *(const v4u*)(abase + (kt << 5));        // ds imm offset kt*64B
      ua.q[1] = *(const v4u*)(abase + (kt << 5) + 16);   // +32B
      const v16bf afr = ua.v;
      __builtin_prefetch(bb + 65536, 0, 1);              // next k-step tile row

      // Group 1: fragments for tiles 0..3 (loads issued together).
      FragU u0[4];
#pragma unroll
      for (int i = 0; i < 4; ++i) {
        const int eo = ((i >> 1) * 32 + (i & 1)) << 9;
        u0[i].q[0] = *(const v4u*)(bb + eo);
        u0[i].q[1] = *(const v4u*)(bb + eo + 8);
      }
      // Group 2: fragments for tiles 4..7 -- in flight during group-1 WMMAs.
      FragU u1[4];
#pragma unroll
      for (int i = 0; i < 4; ++i) {
        const int eo = (((i + 4) >> 1) * 32 + ((i + 4) & 1)) << 9;
        u1[i].q[0] = *(const v4u*)(bb + eo);
        u1[i].q[1] = *(const v4u*)(bb + eo + 8);
      }
#pragma unroll
      for (int i = 0; i < 4; ++i)
        acc[i] = __builtin_amdgcn_wmma_f32_16x16x32_bf16(
            false, afr, false, u0[i].v, (short)0, acc[i], false, false);
#pragma unroll
      for (int i = 0; i < 4; ++i)
        acc[4 + i] = __builtin_amdgcn_wmma_f32_16x16x32_bf16(
            false, afr, false, u1[i].v, (short)0, acc[4 + i], false, false);

      bb += 65536;  // one 64-bit pointer bump per k-step (128 tiles * 512 elems)
    }
    __syncthreads();  // GEMM reads of A done before h/x regions are rewritten

    // LSTM cell + write h (LDS + global y), keep c in regs
    const bool last = (step == T - 1);
#pragma unroll
    for (int j = 0; j < 2; ++j) {
      const v8f vi = acc[0 * 2 + j];
      const v8f vf = acc[1 * 2 + j];
      const v8f vg = acc[2 * 2 + j];
      const v8f vo = acc[3 * 2 + j];
      const int n = ((hbase + j) << 4) + nlo;
#pragma unroll
      for (int r = 0; r < 8; ++r) {
        const int ml = mlB + r;
        const int mg = (mchunk << 4) + ml;
        const float cn = sigm(vf[r]) * creg[j][r] + sigm(vi[r]) * tanhf(vg[r]);
        const float hv = sigm(vo[r]) * tanhf(cn);
        creg[j][r] = cn;
        Als[ml * Ktot + Din + n] = (bf16_t)hv;
        y[(size_t)t * BATCH * yStride + (size_t)mg * yStride + ycol0 + n] = (bf16_t)hv;
        if (last && hT) {
          hT[(size_t)mg * hcStride + ycol0 + n] = hv;
          cT[(size_t)mg * hcStride + ycol0 + n] = cn;
        }
      }
    }

    // Stage next x_t into LDS
    if (step + 1 < T) {
      const int tn = reverse ? (T - 2 - step) : (step + 1);
      const int half = Din >> 1;
      const u32* xs = (const u32*)(xT + (size_t)tn * BATCH * Din) + (size_t)(mchunk * 16) * half;
      for (int idx = tid; idx < 16 * half; idx += 512) {
        const int m = idx / half;
        const int o = idx - m * half;
        *((u32*)(Als + m * Ktot) + o) = xs[(size_t)m * half + o];
      }
    }
    __syncthreads();
  }
}

// ---------------------------------------------------------------------------
// Output projection: out[b][t][f] = dfin[t*64+b][:512] @ out_W^T + out_b
// M=11520 (720 m-tiles), N=128 (8 n-tiles), K=512 (16 k-steps of 32).
// ---------------------------------------------------------------------------
__global__ __launch_bounds__(256)
void out_proj_kernel(const bf16_t* __restrict__ dfin, const bf16_t* __restrict__ Wpk,
                     const float* __restrict__ ob, float* __restrict__ out) {
  const int lane = threadIdx.x & 31;
  const int wave = threadIdx.x >> 5;
  const int tile = blockIdx.x * 8 + wave;   // 720 blocks * 8 waves = 5760 tiles
  const int mtile = tile >> 3;
  const int nt = tile & 7;
  const int nlo = lane & 15;
  const int kb = (lane >> 4) << 3;
  const float bv = ob[(nt << 4) + nlo];
  v8f acc;
#pragma unroll
  for (int r = 0; r < 8; ++r) acc[r] = bv;
  const bf16_t* abase = dfin + (size_t)((mtile << 4) + nlo) * HDIM + kb;
  const bf16_t* bb = Wpk + ((size_t)nt << 9) + (lane << 4);
#pragma unroll 4
  for (int kt = 0; kt < 16; ++kt) {
    FragU ua;
    ua.q[0] = *(const v4u*)(abase + (kt << 5));        // imm offset kt*64B
    ua.q[1] = *(const v4u*)(abase + (kt << 5) + 16);
    FragU ub;
    ub.q[0] = *(const v4u*)(bb + (kt << 12));          // 8 tiles * 512 elems / k-step
    ub.q[1] = *(const v4u*)(bb + (kt << 12) + 8);
    acc = __builtin_amdgcn_wmma_f32_16x16x32_bf16(false, ua.v, false, ub.v, (short)0,
                                                  acc, false, false);
  }
#pragma unroll
  for (int r = 0; r < 8; ++r) {
    const int mg = (mtile << 4) + ((lane >> 4) << 3) + r;
    const int tt = mg >> 6;
    const int bb2 = mg & 63;
    out[((size_t)bb2 * 180 + tt) * 128 + (nt << 4) + nlo] = acc[r];
  }
}

// ---------------------------------------------------------------------------
extern "C" void kernel_launch(void* const* d_in, const int* in_sizes, int n_in,
                              void* d_out, int out_size, void* d_ws, size_t ws_size,
                              hipStream_t stream) {
  (void)in_sizes; (void)n_in; (void)out_size;
  const float* src      = (const float*)d_in[0];
  const float* tgt_in   = (const float*)d_in[1];
  const int*   scenario = (const int*)d_in[2];
  const float* enc_Wih0 = (const float*)d_in[3];
  const float* enc_Whh0 = (const float*)d_in[4];
  const float* enc_b0   = (const float*)d_in[5];
  const float* enc_Wih  = (const float*)d_in[6];
  const float* enc_Whh  = (const float*)d_in[7];
  const float* enc_b    = (const float*)d_in[8];
  const float* h_bW     = (const float*)d_in[9];
  const float* h_bb     = (const float*)d_in[10];
  const float* c_bW     = (const float*)d_in[11];
  const float* c_bb     = (const float*)d_in[12];
  const float* sc_emb   = (const float*)d_in[13];
  const float* dec_Wih0 = (const float*)d_in[14];
  const float* dec_Whh0 = (const float*)d_in[15];
  const float* dec_b0   = (const float*)d_in[16];
  const float* dec_Wih  = (const float*)d_in[17];
  const float* dec_Whh  = (const float*)d_in[18];
  const float* dec_b    = (const float*)d_in[19];
  const float* out_W    = (const float*)d_in[20];
  const float* out_b    = (const float*)d_in[21];
  float* out = (float*)d_out;

  char* ws = (char*)d_ws;
  size_t off = 0;
  auto take = [&](size_t bytes) -> char* {
    char* p = ws + off;
    off += (bytes + 255) & ~(size_t)255;
    return p;
  };

  bf16_t* x0   = (bf16_t*)take((size_t)300 * 64 * 128 * 2);
  bf16_t* bufA = (bf16_t*)take((size_t)300 * 64 * 1024 * 2);
  bf16_t* bufB = (bf16_t*)take((size_t)300 * 64 * 1024 * 2);
  bf16_t* d0   = (bf16_t*)take((size_t)180 * 64 * 640 * 2);
  bf16_t* dA   = (bf16_t*)take((size_t)180 * 64 * 512 * 2);
  bf16_t* dB   = (bf16_t*)take((size_t)180 * 64 * 512 * 2);
  float*  hcat = (float*)take((size_t)4 * 64 * 1024 * 4);
  float*  ccat = (float*)take((size_t)4 * 64 * 1024 * 4);
  float*  h0d  = (float*)take((size_t)4 * 64 * 512 * 4);
  float*  c0d  = (float*)take((size_t)4 * 64 * 512 * 4);
  bf16_t* Wenc[4][2];
  for (int l = 0; l < 4; ++l)
    for (int d = 0; d < 2; ++d) {
      const size_t elems = (size_t)((l == 0 ? 640 : 1536) / 32) * 128 * 512;
      Wenc[l][d] = (bf16_t*)take(elems * 2);
    }
  bf16_t* Wdec[4];
  for (int l = 0; l < 4; ++l) {
    const size_t elems = (size_t)((l == 0 ? 1152 : 1024) / 32) * 128 * 512;
    Wdec[l] = (bf16_t*)take(elems * 2);
  }
  bf16_t* Wout = (bf16_t*)take((size_t)16 * 8 * 512 * 2);
  if (off > ws_size) return;  // insufficient scratch

  // --- Pack weights (bf16, WMMA fragment order) ---
  for (int l = 0; l < 4; ++l) {
    const int Din = (l == 0) ? 128 : 1024;
    const int Ktot = Din + 512;
    const size_t total = (size_t)(Ktot / 32) * 128 * 512;
    for (int d = 0; d < 2; ++d) {
      const float* Wih_p = (l == 0) ? enc_Wih0 + (size_t)d * 2048 * 128
                                    : enc_Wih + (size_t)((l - 1) * 2 + d) * 2048 * 1024;
      const float* Whh_p = (l == 0) ? enc_Whh0 + (size_t)d * 2048 * 512
                                    : enc_Whh + (size_t)((l - 1) * 2 + d) * 2048 * 512;
      pack_w_kernel<<<(unsigned)(total / 256), 256, 0, stream>>>(Wih_p, Whh_p, Wenc[l][d],
                                                                 Din, Ktot, 2048);
    }
  }
  for (int l = 0; l < 4; ++l) {
    const int Din = (l == 0) ? 640 : 512;
    const int Ktot = Din + 512;
    const size_t total = (size_t)(Ktot / 32) * 128 * 512;
    const float* Wih_p = (l == 0) ? dec_Wih0 : dec_Wih + (size_t)(l - 1) * 2048 * 512;
    const float* Whh_p = (l == 0) ? dec_Whh0 : dec_Whh + (size_t)(l - 1) * 2048 * 512;
    pack_w_kernel<<<(unsigned)(total / 256), 256, 0, stream>>>(Wih_p, Whh_p, Wdec[l],
                                                               Din, Ktot, 2048);
  }
  pack_w_kernel<<<(16 * 8 * 512) / 256, 256, 0, stream>>>(out_W, out_W, Wout, 512, 512, 128);

  // --- Activations to bf16 time-major ---
  to_tm_bf16<<<(300 * 64 * 128) / 256, 256, 0, stream>>>(src, x0, 64, 300, 128);
  build_dec_in<<<(180 * 64 * 640) / 256, 256, 0, stream>>>(tgt_in, scenario, sc_emb, d0, 180);

  // --- Encoder: 4 bidirectional layers ---
  lstm_layer_kernel<<<dim3(4, 2), 512, 0, stream>>>(
      x0, 300, 128, Wenc[0][0], Wenc[0][1], enc_b0, enc_b0 + 2048,
      bufA, 1024, 1, nullptr, nullptr, hcat + 0, ccat + 0, 1024);
  lstm_layer_kernel<<<dim3(4, 2), 512, 0, stream>>>(
      bufA, 300, 1024, Wenc[1][0], Wenc[1][1], enc_b + 0 * 2 * 2048, enc_b + (0 * 2 + 1) * 2048,
      bufB, 1024, 1, nullptr, nullptr, hcat + (size_t)1 * 64 * 1024, ccat + (size_t)1 * 64 * 1024, 1024);
  lstm_layer_kernel<<<dim3(4, 2), 512, 0, stream>>>(
      bufB, 300, 1024, Wenc[2][0], Wenc[2][1], enc_b + 1 * 2 * 2048, enc_b + (1 * 2 + 1) * 2048,
      bufA, 1024, 1, nullptr, nullptr, hcat + (size_t)2 * 64 * 1024, ccat + (size_t)2 * 64 * 1024, 1024);
  lstm_layer_kernel<<<dim3(4, 2), 512, 0, stream>>>(
      bufA, 300, 1024, Wenc[3][0], Wenc[3][1], enc_b + 2 * 2 * 2048, enc_b + (2 * 2 + 1) * 2048,
      bufB, 1024, 1, nullptr, nullptr, hcat + (size_t)3 * 64 * 1024, ccat + (size_t)3 * 64 * 1024, 1024);

  // --- Bridge ---
  bridge_kernel<<<512, 256, 0, stream>>>(hcat, ccat, h_bW, h_bb, c_bW, c_bb, h0d, c0d);

  // --- Decoder: 4 unidirectional layers with bridged initial states ---
  lstm_layer_kernel<<<dim3(4, 1), 512, 0, stream>>>(
      d0, 180, 640, Wdec[0], Wdec[0], dec_b0, dec_b0,
      dA, 512, 0, h0d + (size_t)0 * 64 * 512, c0d + (size_t)0 * 64 * 512, nullptr, nullptr, 512);
  lstm_layer_kernel<<<dim3(4, 1), 512, 0, stream>>>(
      dA, 180, 512, Wdec[1], Wdec[1], dec_b + 0 * 2048, dec_b + 0 * 2048,
      dB, 512, 0, h0d + (size_t)1 * 64 * 512, c0d + (size_t)1 * 64 * 512, nullptr, nullptr, 512);
  lstm_layer_kernel<<<dim3(4, 1), 512, 0, stream>>>(
      dB, 180, 512, Wdec[2], Wdec[2], dec_b + 1 * 2048, dec_b + 1 * 2048,
      dA, 512, 0, h0d + (size_t)2 * 64 * 512, c0d + (size_t)2 * 64 * 512, nullptr, nullptr, 512);
  lstm_layer_kernel<<<dim3(4, 1), 512, 0, stream>>>(
      dA, 180, 512, Wdec[3], Wdec[3], dec_b + 2 * 2048, dec_b + 2 * 2048,
      dB, 512, 0, h0d + (size_t)3 * 64 * 512, c0d + (size_t)3 * 64 * 512, nullptr, nullptr, 512);

  // --- Output projection ---
  out_proj_kernel<<<720, 256, 0, stream>>>(dB, Wout, out_b, out);
}
